// BitLinear_29197187678258
// MI455X (gfx1250) — 8-bit matrix operand order repaired and checked by reading, compiled, not run on hardware
//
#include <hip/hip_runtime.h>
#include <hip/hip_bf16.h>
#include <stddef.h>

typedef __attribute__((ext_vector_type(8))) int v8i;

#define TILE_M 256
#define TILE_N 128
#define TILE_K 64
#define LDS_ROW_I 20

__device__ __forceinline__ void async_load_b128(void* lds_ptr, const void* gptr) {
    unsigned lds_off = (unsigned)(size_t)lds_ptr;
    asm volatile("global_load_async_to_lds_b128 %0, %1, off"
                 :: "v"(lds_off), "v"(gptr) : "memory");
}

__device__ __forceinline__ void wait_async_le6() {
#if __has_builtin(__builtin_amdgcn_s_wait_asynccnt)
    __builtin_amdgcn_s_wait_asynccnt(6);
#else
    asm volatile("s_wait_asynccnt 0x6" ::: "memory");
#endif
}

__device__ __forceinline__ void wait_async_le0() {
#if __has_builtin(__builtin_amdgcn_s_wait_asynccnt)
    __builtin_amdgcn_s_wait_asynccnt(0);
#else
    asm volatile("s_wait_asynccnt 0x0" ::: "memory");
#endif
}

__global__ __launch_bounds__(256)
void absum_part(const float4* __restrict__ w, float* __restrict__ partial, int n4) {
    int idx = blockIdx.x * 256 + threadIdx.x;
    int stride = gridDim.x * 256;
    float s = 0.f;
    for (int i = idx; i < n4; i += stride) {
        float4 v = w[i];
        s += fabsf(v.x) + fabsf(v.y) + fabsf(v.z) + fabsf(v.w);
    }
    __shared__ float red[256];
    red[threadIdx.x] = s;
    __syncthreads();
    for (int off = 128; off > 0; off >>= 1) {
        if (threadIdx.x < off) red[threadIdx.x] += red[threadIdx.x + off];
        __syncthreads();
    }
    if (threadIdx.x == 0) partial[blockIdx.x] = red[0];
}

__global__ __launch_bounds__(256)
void absum_final(const float* __restrict__ partial, float* __restrict__ thresh,
                 int n, float inv_count) {
    float s = 0.f;
    for (int i = threadIdx.x; i < n; i += 256) s += partial[i];
    __shared__ float red[256];
    red[threadIdx.x] = s;
    __syncthreads();
    for (int off = 128; off > 0; off >>= 1) {
        if (threadIdx.x < off) red[threadIdx.x] += red[threadIdx.x + off];
        __syncthreads();
    }
    if (threadIdx.x == 0) thresh[0] = red[0] * inv_count;
}

__device__ inline unsigned pack4_q(float4 v, float s) {
    int a = (int)__builtin_rintf(v.x * s);
    int b = (int)__builtin_rintf(v.y * s);
    int c = (int)__builtin_rintf(v.z * s);
    int d = (int)__builtin_rintf(v.w * s);
    return (unsigned)(a & 255) | ((unsigned)(b & 255) << 8) |
           ((unsigned)(c & 255) << 16) | ((unsigned)(d & 255) << 24);
}

__global__ __launch_bounds__(256)
void quant_x(const float* __restrict__ x, signed char* __restrict__ q,
             float* __restrict__ dq, int K) {
    const int row = blockIdx.x;
    const int t = threadIdx.x;
    const float4* xr = (const float4*)(x + (size_t)row * K);
    float4 a = xr[2 * t];
    float4 b = xr[2 * t + 1];
    float m = fmaxf(fmaxf(fmaxf(fabsf(a.x), fabsf(a.y)), fmaxf(fabsf(a.z), fabsf(a.w))),
                    fmaxf(fmaxf(fabsf(b.x), fabsf(b.y)), fmaxf(fabsf(b.z), fabsf(b.w))));
    __shared__ float red[256];
    red[t] = m;
    __syncthreads();
    for (int off = 128; off > 0; off >>= 1) {
        if (t < off) red[t] = fmaxf(red[t], red[t + off]);
        __syncthreads();
    }
    float xmax = red[0];
    float scale = 127.0f / (xmax + 1e-8f);
    uint2 pk;
    pk.x = pack4_q(a, scale);
    pk.y = pack4_q(b, scale);
    ((uint2*)(q + (size_t)row * K))[t] = pk;
    if (t == 0) dq[row] = (xmax + 1e-8f) / 127.0f;
}

__device__ inline int tern1(float w, float th) {
    return (fabsf(w) > th) ? (w > 0.f ? 1 : -1) : 0;
}

__global__ __launch_bounds__(256)
void ternarize(const float4* __restrict__ w, const float* __restrict__ thresh,
               unsigned* __restrict__ t, int n4) {
    const float th = thresh[0];
    int idx = blockIdx.x * 256 + threadIdx.x;
    int stride = gridDim.x * 256;
    for (int i = idx; i < n4; i += stride) {
        float4 v = w[i];
        unsigned p = (unsigned)(tern1(v.x, th) & 255) |
                     ((unsigned)(tern1(v.y, th) & 255) << 8) |
                     ((unsigned)(tern1(v.z, th) & 255) << 16) |
                     ((unsigned)(tern1(v.w, th) & 255) << 24);
        t[i] = p;
    }
}

__global__ __launch_bounds__(256)
void gemm_iu8(const signed char* __restrict__ Aq, const signed char* __restrict__ Bt,
              const float* __restrict__ dq, const float* __restrict__ wscale_p,
              float* __restrict__ out, int M, int N, int K) {
    __shared__ int lA[2][TILE_M * LDS_ROW_I];
    __shared__ int lB[2][TILE_N * LDS_ROW_I];

    const int tid   = threadIdx.x;
    const int lane  = tid & 31;
    const int wave  = tid >> 5;
    const int waveM = wave >> 1;
    const int waveN = wave & 1;
    const int l15   = lane & 15;
    const int h     = lane >> 4;

    const int m0 = blockIdx.y * TILE_M;
    const int n0 = blockIdx.x * TILE_N;

    const int ch   = tid & 3;
    const int rbase = tid >> 2;
    const signed char* gA = Aq + (size_t)(m0 + rbase) * K + ch * 16;
    const signed char* gB = Bt + (size_t)(n0 + rbase) * K + ch * 16;

    v8i acc[4][4];
#pragma unroll
    for (int i = 0; i < 4; ++i)
#pragma unroll
        for (int j = 0; j < 4; ++j) {
            v8i z = {0, 0, 0, 0, 0, 0, 0, 0};
            acc[i][j] = z;
        }

#pragma unroll
    for (int i = 0; i < 4; ++i)
        async_load_b128(&lA[0][(rbase + i * 64) * LDS_ROW_I + ch * 4],
                        gA + (size_t)(i * 64) * K);
#pragma unroll
    for (int i = 0; i < 2; ++i)
        async_load_b128(&lB[0][(rbase + i * 64) * LDS_ROW_I + ch * 4],
                        gB + (size_t)(i * 64) * K);

    int cur = 0;
    for (int k0 = 0; k0 < K; k0 += TILE_K) {
        const bool has_next = (k0 + TILE_K) < K;
        if (has_next) {
            const int nb = cur ^ 1;
            const size_t knext = (size_t)(k0 + TILE_K);
#pragma unroll
            for (int i = 0; i < 4; ++i)
                async_load_b128(&lA[nb][(rbase + i * 64) * LDS_ROW_I + ch * 4],
                                gA + (size_t)(i * 64) * K + knext);
#pragma unroll
            for (int i = 0; i < 2; ++i)
                async_load_b128(&lB[nb][(rbase + i * 64) * LDS_ROW_I + ch * 4],
                                gB + (size_t)(i * 64) * K + knext);
            wait_async_le6();
        } else {
            wait_async_le0();
        }
        __syncthreads();

        v8i bfrag[4];
#pragma unroll
        for (int nt = 0; nt < 4; ++nt) {
            const int* bp = &lB[cur][(waveN * 64 + nt * 16 + l15) * LDS_ROW_I];
            int o = h * 4;
            v8i b;
            b[0] = bp[o + 0]; b[1] = bp[o + 1]; b[2] = bp[o + 2]; b[3] = bp[o + 3];
            b[4] = bp[o + 8]; b[5] = bp[o + 9]; b[6] = bp[o + 10]; b[7] = bp[o + 11];
            bfrag[nt] = b;
        }

#pragma unroll
        for (int mt = 0; mt < 4; ++mt) {
            const int* ap = &lA[cur][(waveM * 64 + mt * 16 + l15) * LDS_ROW_I];
            int o = h * 4;
            v8i a;
            a[0] = ap[o + 0];  a[1] = ap[o + 1];
            a[2] = ap[o + 2];  a[3] = ap[o + 3];
            a[4] = ap[o + 8];  a[5] = ap[o + 9];
            a[6] = ap[o + 10]; a[7] = ap[o + 11];
#pragma unroll
            for (int nt = 0; nt < 4; ++nt) {
                acc[mt][nt] = __builtin_amdgcn_wmma_i32_16x16x64_iu8(
                     true, a,  true, bfrag[nt],
                    acc[mt][nt],  false,  false);
            }
        }
        __syncthreads();
        cur ^= 1;
    }

    const float ws = wscale_p[0];
#pragma unroll
    for (int mt = 0; mt < 4; ++mt) {
        int mbase = m0 + waveM * 64 + mt * 16 + h * 8;
#pragma unroll
        for (int nt = 0; nt < 4; ++nt) {
            int n = n0 + waveN * 64 + nt * 16 + l15;
#pragma unroll
            for (int j = 0; j < 8; ++j) {
                int m = mbase + j;
                out[(size_t)m * N + n] = (float)acc[mt][nt][j] * dq[m] * ws;
            }
        }
    }
}

extern "C" void kernel_launch(void* const* d_in, const int* in_sizes, int n_in,
                              void* d_out, int out_size, void* d_ws, size_t ws_size,
                              hipStream_t stream) {
    const float* x      = (const float*)d_in[0];
    const float* w      = (const float*)d_in[1];
    const float* wscale = (const float*)d_in[2];

    const int K = 2048;
    const int M = in_sizes[0] / K;
    const int N = in_sizes[1] / K;

    signed char* Aq = (signed char*)d_ws;
    signed char* Bt = Aq + (size_t)M * K;
    float* dq       = (float*)(Bt + (size_t)N * K);
    float* partial  = dq + M;
    float* thresh   = partial + 1024;

    int wn4 = (N * K) / 4;
    absum_part<<<1024, 256, 0, stream>>>((const float4*)w, partial, wn4);
    absum_final<<<1, 256, 0, stream>>>(partial, thresh, 1024,
                                       1.0f / ((float)N * (float)K));
    quant_x<<<M, 256, 0, stream>>>(x, Aq, dq, K);
    ternarize<<<4096, 256, 0, stream>>>((const float4*)w, thresh, (unsigned*)Bt, wn4);

    dim3 grid(N / TILE_N, M / TILE_M);
    gemm_iu8<<<grid, 256, 0, stream>>>(Aq, Bt, dq, wscale, (float*)d_out, M, N, K);
}
